// PlasmidLMAttention_19035295056478
// MI455X (gfx1250) — compile-verified, matrix operand order checked by reading, not run on hardware
//
#include <hip/hip_runtime.h>

// ---------------- problem constants (B=2, S=2048, D=2048, H=16, hd=128) ----
#define NB 2
#define NS 2048
#define ND 2048
#define NH 16
#define HD 128
#define NT (NB * NS)            // 4096 tokens

typedef __bf16 v16bf __attribute__((ext_vector_type(16)));
typedef float  v8f   __attribute__((ext_vector_type(8)));
typedef unsigned int u32x4 __attribute__((ext_vector_type(4)));
typedef int          i32x4 __attribute__((ext_vector_type(4)));
typedef int          i32x8 __attribute__((ext_vector_type(8)));

union FragAB {                   // 16 bf16 = one WMMA A/B fragment per lane
    v16bf v;
    uint4 u[2];
};

__device__ __forceinline__ unsigned short f2bf(float f) {
    unsigned u = __float_as_uint(f);
    unsigned r = u + 0x7FFFu + ((u >> 16) & 1u);   // round-to-nearest-even
    return (unsigned short)(r >> 16);
}

// ------------------------- Tensor Data Mover (gfx1250) --------------------
#if __has_builtin(__builtin_amdgcn_tensor_load_to_lds)
  #define TDM_AVAILABLE 1
  #if __has_include(<hip/amd_detail/amd_gfx1250_TDM.h>)
    #define TDM_SIX_ARGS 1          // amdgpu-toolchain / therock-10.0 headers
    #warning "CDNA5 TDM path: tensor_load_to_lds (6-arg form) enabled"
  #else
    #define TDM_SIX_ARGS 0          // ROCm 7.2 5-arg form
    #warning "CDNA5 TDM path: tensor_load_to_lds (5-arg form) enabled"
  #endif
#else
  #define TDM_AVAILABLE 0
  #warning "TDM builtin NOT available - cooperative global->LDS copy fallback"
#endif

#if TDM_AVAILABLE
// 2D tile load: tile_d0 contiguous elements (bf16) x tile_d1 rows with
// row stride stride_d0 (elements).  D# per CDNA5 ISA.
__device__ __forceinline__ void tdm_load_2d(unsigned lds_addr, const void* gptr,
                                            unsigned tile_d0, unsigned tile_d1,
                                            unsigned long long stride_d0)
{
    unsigned long long ga = (unsigned long long)gptr;
    u32x4 g0 = { 1u,                                   // count=1, user mode
                 lds_addr,                             // D#.lds_addr
                 (unsigned)ga,                         // global_addr[31:0]
                 (unsigned)(ga >> 32) | 0x80000000u }; // addr[56:32] | type=2
    i32x8 g1 = {
        0x00010000,                                               // data_size=2B
        (int)((tile_d0 & 0xFFFFu) << 16),                         // tensor_dim0 lo
        (int)(((tile_d0 >> 16) & 0xFFFFu) | ((tile_d1 & 0xFFFFu) << 16)),
        (int)(((tile_d1 >> 16) & 0xFFFFu) | ((tile_d0 & 0xFFFFu) << 16)), // tile_dim0
        (int)(tile_d1 & 0xFFFFu),                                 // tile_dim1
        (int)(unsigned)stride_d0,                                 // dim0 stride lo
        (int)((stride_d0 >> 32) & 0xFFFFu),                       // stride hi
        0 };
    i32x4 gz = { 0, 0, 0, 0 };
#if TDM_SIX_ARGS
    i32x8 gz8 = { 0, 0, 0, 0, 0, 0, 0, 0 };
    __builtin_amdgcn_tensor_load_to_lds(g0, g1, gz, gz, gz8, 0);
#else
    __builtin_amdgcn_tensor_load_to_lds(g0, g1, gz, gz, 0);
#endif
}
#endif

// ---------------------------------------------------------------- convert --
__global__ void k_f32_to_bf16(const float* __restrict__ src,
                              unsigned short* __restrict__ dst, int n) {
    int i = blockIdx.x * blockDim.x + threadIdx.x;
    int stride = gridDim.x * blockDim.x;
    for (; i < n; i += stride) dst[i] = f2bf(src[i]);
}

// ------------------------------------------------------------------- GEMM --
// Y = A(4096 x 2048, bf16 row-major) @ W^T (W: 2048 x 2048 bf16 row-major).
// One wave computes a 32(M) x 64(N) tile: 2x4 accumulators, K-step 32; each
// B fragment is reused by both M-subtiles (compute/load ratio 8 WMMA : 12
// b128).  Fragments for k+1 are prefetched while k's WMMAs issue.
// mode 0: f32 -> (B,H,S,hd) layout (Q/K pre-RoPE)
// mode 1: mode 0 + bf16 transposed (B,H,hd,S) copy (V)
// mode 2: f32 row-major (token, D) (final output projection)
__global__ __launch_bounds__(128)
void k_gemm_bf16(const unsigned short* __restrict__ A,
                 const unsigned short* __restrict__ W,
                 float* __restrict__ out_f32,
                 unsigned short* __restrict__ out_bf16_t,
                 int mode)
{
    const int lane  = threadIdx.x & 31;
    const int wave  = threadIdx.x >> 5;
    const int gw    = blockIdx.x * 4 + wave;     // 0..4095
    const int mt    = gw >> 5;                   // 128 M-tiles of 32
    const int nt    = gw & 31;                   // 32 N-tiles of 64
    const int m0    = mt * 32;
    const int n0    = nt * 64;
    const int lrow  = lane & 15;
    const int khalf = (lane >> 4) * 8;           // 0 or 8 (A/B layout halves)

    v8f zv = {};
    v8f acc[2][4];
    #pragma unroll
    for (int mi = 0; mi < 2; ++mi)
        #pragma unroll
        for (int j = 0; j < 4; ++j) acc[mi][j] = zv;

    const unsigned short* ar[2] = { A + (size_t)(m0 + lrow) * ND,
                                    A + (size_t)(m0 + 16 + lrow) * ND };
    const unsigned short* wr[4];
    #pragma unroll
    for (int j = 0; j < 4; ++j) wr[j] = W + (size_t)(n0 + j * 16 + lrow) * ND;

    FragAB fa[2], fb[4];
    #pragma unroll
    for (int mi = 0; mi < 2; ++mi) {
        fa[mi].u[0] = *(const uint4*)(ar[mi] + khalf);
        fa[mi].u[1] = *(const uint4*)(ar[mi] + khalf + 16);
    }
    #pragma unroll
    for (int j = 0; j < 4; ++j) {
        fb[j].u[0] = *(const uint4*)(wr[j] + khalf);
        fb[j].u[1] = *(const uint4*)(wr[j] + khalf + 16);
    }

    const int NITER = ND / 32;
    #pragma unroll 2
    for (int it = 0; it < NITER - 1; ++it) {
        const int off = (it + 1) * 32 + khalf;
        FragAB na[2], nb[4];
        #pragma unroll
        for (int mi = 0; mi < 2; ++mi) {
            na[mi].u[0] = *(const uint4*)(ar[mi] + off);
            na[mi].u[1] = *(const uint4*)(ar[mi] + off + 16);
        }
        __builtin_prefetch(ar[0] + off + 64, 0, 1);   // global_prefetch_b8
        #pragma unroll
        for (int j = 0; j < 4; ++j) {
            nb[j].u[0] = *(const uint4*)(wr[j] + off);
            nb[j].u[1] = *(const uint4*)(wr[j] + off + 16);
        }
        #pragma unroll
        for (int j = 0; j < 4; ++j)
            #pragma unroll
            for (int mi = 0; mi < 2; ++mi)
                acc[mi][j] = __builtin_amdgcn_wmma_f32_16x16x32_bf16(
                    false, fa[mi].v, false, fb[j].v, (short)0, acc[mi][j],
                    false, false);
        #pragma unroll
        for (int mi = 0; mi < 2; ++mi) fa[mi] = na[mi];
        #pragma unroll
        for (int j = 0; j < 4; ++j) fb[j] = nb[j];
    }
    #pragma unroll
    for (int j = 0; j < 4; ++j)
        #pragma unroll
        for (int mi = 0; mi < 2; ++mi)
            acc[mi][j] = __builtin_amdgcn_wmma_f32_16x16x32_bf16(
                false, fa[mi].v, false, fb[j].v, (short)0, acc[mi][j],
                false, false);

    #pragma unroll
    for (int mi = 0; mi < 2; ++mi) {
        #pragma unroll
        for (int r = 0; r < 8; ++r) {
            const int row = m0 + mi * 16 + r + ((lane >> 4) << 3); // token idx
            const int b = row >> 11;                               // / 2048
            const int s = row & 2047;
            #pragma unroll
            for (int j = 0; j < 4; ++j) {
                const int n = n0 + j * 16 + lrow;
                const float val = acc[mi][j][r];
                if (mode == 2) {
                    out_f32[(size_t)row * ND + n] = val;
                } else {
                    const int h  = n >> 7;                         // / 128
                    const int d  = n & 127;
                    const int bh = b * NH + h;
                    out_f32[((size_t)bh * NS + s) * HD + d] = val;
                    if (mode == 1)
                        out_bf16_t[((size_t)bh * HD + d) * NS + s] = f2bf(val);
                }
            }
        }
    }
}

// ------------------------------------------------------------------- RoPE --
__global__ void k_rope(const float* __restrict__ Qf, const float* __restrict__ Kf,
                       const float* __restrict__ cosT, const float* __restrict__ sinT,
                       unsigned short* __restrict__ Qbf,
                       float* __restrict__ Kout,
                       unsigned short* __restrict__ Kbf)
{
    const int NP = NB * NH * NS * (HD / 2);
    int i = blockIdx.x * blockDim.x + threadIdx.x;
    if (i >= NP) return;
    const int d2 = i & 63;
    const int s  = (i >> 6) & 2047;
    const int bh = i >> 17;
    const float c  = cosT[s * 64 + d2];
    const float sn = sinT[s * 64 + d2];
    const size_t base = ((size_t)bh * NS + s) * HD + 2 * d2;
    {
        float x1 = Qf[base], x2 = Qf[base + 1];
        Qbf[base]     = f2bf(x1 * c - x2 * sn);
        Qbf[base + 1] = f2bf(x1 * sn + x2 * c);
    }
    {
        float x1 = Kf[base], x2 = Kf[base + 1];
        float y1 = x1 * c - x2 * sn;
        float y2 = x1 * sn + x2 * c;
        Kout[base] = y1;  Kout[base + 1] = y2;
        Kbf[base]  = f2bf(y1);
        Kbf[base + 1] = f2bf(y2);
    }
}

// -------------------------------------------------------- flash attention --
// One workgroup (4 waves) owns 64 consecutive queries of one (b,h); each wave
// has a 16-query tile.  K/V kv-tiles (32 rows) are staged to double-buffered
// LDS — via the Tensor Data Mover when available (tensor_load_to_lds +
// s_wait_tensorcnt), else a cooperative copy — and shared by all 4 waves.
__global__ __launch_bounds__(128)
void k_flash_attn(const unsigned short* __restrict__ Qbf,
                  const unsigned short* __restrict__ Kbf,
                  const unsigned short* __restrict__ Vt,
                  unsigned short* __restrict__ Ctx)
{
    __shared__ __align__(16) unsigned short ldsK[2][32 * HD];  // 2 x 8 KB
    __shared__ __align__(16) unsigned short ldsV[2][HD * 32];  // 2 x 8 KB
    __shared__ __align__(16) unsigned short ldsP[4][16 * 32];  // 4 x 1 KB

    const int lane  = threadIdx.x & 31;
    const int wave  = threadIdx.x >> 5;
    const int bh    = blockIdx.x >> 5;          // 32 q-blocks per (b,h)
    const int qblk  = blockIdx.x & 31;
    const int b     = bh >> 4;
    const int h     = bh & 15;
    const int qB    = qblk * 64;                // block's first query
    const int q0    = qB + wave * 16;           // wave's first query
    const int qmax  = qB + 63;
    const int lrow  = lane & 15;
    const int khalf = (lane >> 4) * 8;
    const float scale = 0.08838834764831845f;   // 1/sqrt(128)

    const unsigned short* kb_ptr = Kbf + (size_t)bh * NS * HD;
    const unsigned short* vb_ptr = Vt  + (size_t)bh * HD * NS;
    unsigned short* plds = ldsP[wave];

    // Preload the 4 Q fragments (K-dim 128 = 4 x 32)
    FragAB qf[4];
    const unsigned short* qrow = Qbf + ((size_t)bh * NS + q0 + lrow) * HD;
    #pragma unroll
    for (int d = 0; d < 4; ++d) {
        const int off = d * 32 + khalf;
        qf[d].u[0] = *(const uint4*)(qrow + off);
        qf[d].u[1] = *(const uint4*)(qrow + off + 16);
    }

    v8f zv = {};
    v8f o[8];
    #pragma unroll
    for (int t = 0; t < 8; ++t) o[t] = zv;
    float mrow[8], lsum[8], alphas[8];
    #pragma unroll
    for (int r = 0; r < 8; ++r) { mrow[r] = -1e30f; lsum[r] = 0.0f; }

    // ---- staging helper -------------------------------------------------
    auto stage = [&](int kb, int bsel) {
#if TDM_AVAILABLE
        if (wave == 0) {
            tdm_load_2d((unsigned)(size_t)(void*)&ldsK[bsel][0],
                        kb_ptr + (size_t)kb * HD, HD, 32, HD);
            tdm_load_2d((unsigned)(size_t)(void*)&ldsV[bsel][0],
                        vb_ptr + kb, 32, HD, NS);
        }
#else
        const uint4* gk = (const uint4*)(kb_ptr + (size_t)kb * HD);
        uint4* lk = (uint4*)&ldsK[bsel][0];
        for (int i = threadIdx.x; i < 512; i += 128) lk[i] = gk[i];
        const unsigned short* gv = vb_ptr + kb;
        uint4* lv = (uint4*)&ldsV[bsel][0];
        for (int i = threadIdx.x; i < 512; i += 128) {
            const int row = i >> 2, col = i & 3;
            lv[i] = *(const uint4*)(gv + (size_t)row * NS + col * 8);
        }
#endif
    };

    stage(0, 0);
    int buf = 0;
    for (int kb = 0; kb <= qmax; kb += 32, buf ^= 1) {
#if TDM_AVAILABLE
        if (wave == 0) __builtin_amdgcn_s_wait_tensorcnt(0);
#endif
        __syncthreads();                       // buf tile visible to all waves
        if (kb + 32 <= qmax) stage(kb + 32, buf ^ 1);

        if (kb <= q0 + 15) {
            const unsigned short* lk = &ldsK[buf][0];
            const unsigned short* lv = &ldsV[buf][0];

            // ---- scores: 16q x 32k, 4 K-steps over hd -------------------
            v8f s0 = zv, s1 = zv;
            #pragma unroll
            for (int d = 0; d < 4; ++d) {
                const int off = d * 32 + khalf;
                const unsigned short* kr0 = lk + (lrow)      * HD + off;
                const unsigned short* kr1 = lk + (16 + lrow) * HD + off;
                FragAB kf0, kf1;
                kf0.u[0] = *(const uint4*)(kr0);  kf0.u[1] = *(const uint4*)(kr0 + 16);
                kf1.u[0] = *(const uint4*)(kr1);  kf1.u[1] = *(const uint4*)(kr1 + 16);
                s0 = __builtin_amdgcn_wmma_f32_16x16x32_bf16(
                    false, qf[d].v, false, kf0.v, (short)0, s0, false, false);
                s1 = __builtin_amdgcn_wmma_f32_16x16x32_bf16(
                    false, qf[d].v, false, kf1.v, (short)0, s1, false, false);
            }

            // ---- online softmax over the 32 new columns -----------------
            #pragma unroll
            for (int r = 0; r < 8; ++r) {
                const int row = q0 + r + ((lane >> 4) << 3);
                const int c0  = kb + lrow;
                const int c1  = c0 + 16;
                float a0 = (c0 <= row) ? s0[r] * scale : -1e9f;
                float a1 = (c1 <= row) ? s1[r] * scale : -1e9f;
                float mx = fmaxf(a0, a1);
                mx = fmaxf(mx, __shfl_xor(mx, 1, 32));
                mx = fmaxf(mx, __shfl_xor(mx, 2, 32));
                mx = fmaxf(mx, __shfl_xor(mx, 4, 32));
                mx = fmaxf(mx, __shfl_xor(mx, 8, 32));
                const float mnew  = fmaxf(mrow[r], mx);
                const float alpha = __expf(mrow[r] - mnew);
                const float p0 = __expf(a0 - mnew);
                const float p1 = __expf(a1 - mnew);
                float rs = p0 + p1;
                rs += __shfl_xor(rs, 1, 32);
                rs += __shfl_xor(rs, 2, 32);
                rs += __shfl_xor(rs, 4, 32);
                rs += __shfl_xor(rs, 8, 32);
                lsum[r] = lsum[r] * alpha + rs;
                mrow[r] = mnew;
                alphas[r] = alpha;
                const int prow = r + ((lane >> 4) << 3);
                plds[prow * 32 + lrow]      = f2bf(p0);   // C-layout -> LDS
                plds[prow * 32 + 16 + lrow] = f2bf(p1);
            }
            asm volatile("s_wait_dscnt 0x0" ::: "memory");

            // reload P as an A-fragment (16 x 32 bf16)
            FragAB pf;
            {
                const int base = lrow * 32 + khalf;
                pf.u[0] = *(const uint4*)(plds + base);
                pf.u[1] = *(const uint4*)(plds + base + 16);
            }

            // ---- O = diag(alpha)*O + P @ V ------------------------------
            #pragma unroll
            for (int t = 0; t < 8; ++t) {
                v8f ot = o[t];
                #pragma unroll
                for (int r = 0; r < 8; ++r) ot[r] *= alphas[r];
                const unsigned short* vr = lv + (t * 16 + lrow) * 32 + khalf;
                FragAB vf;
                vf.u[0] = *(const uint4*)(vr);
                vf.u[1] = *(const uint4*)(vr + 16);
                o[t] = __builtin_amdgcn_wmma_f32_16x16x32_bf16(
                    false, pf.v, false, vf.v, (short)0, ot, false, false);
            }
        }
        __syncthreads();                       // done reading buf
    }

    // ---- epilogue: normalize, store bf16 context (token-major B,S,D) ----
    #pragma unroll
    for (int r = 0; r < 8; ++r) {
        const int rowg  = q0 + r + ((lane >> 4) << 3);
        const int token = b * NS + rowg;
        const float inv = 1.0f / lsum[r];
        #pragma unroll
        for (int t = 0; t < 8; ++t) {
            const int n = h * HD + t * 16 + lrow;
            Ctx[(size_t)token * ND + n] = f2bf(o[t][r] * inv);
        }
    }
}

// ----------------------------------------------------------------- launch --
extern "C" void kernel_launch(void* const* d_in, const int* in_sizes, int n_in,
                              void* d_out, int out_size, void* d_ws, size_t ws_size,
                              hipStream_t stream) {
    (void)in_sizes; (void)n_in; (void)out_size; (void)ws_size;

    const float* hidden = (const float*)d_in[0];
    const float* cosT   = (const float*)d_in[1];
    const float* sinT   = (const float*)d_in[2];
    const float* wq     = (const float*)d_in[3];
    const float* wk     = (const float*)d_in[4];
    const float* wv     = (const float*)d_in[5];
    const float* wo     = (const float*)d_in[6];

    float* out  = (float*)d_out;                        // (B,S,D)
    float* kout = out  + (size_t)NB * NS * ND;          // (B,H,S,hd)
    float* vout = kout + (size_t)NB * NH * NS * HD;     // (B,H,S,hd)

    char* ws = (char*)d_ws;
    unsigned short* Xbf = (unsigned short*)ws;  ws += (size_t)NT * ND * 2;
    unsigned short* Wqb = (unsigned short*)ws;  ws += (size_t)ND * ND * 2;
    unsigned short* Wkb = (unsigned short*)ws;  ws += (size_t)ND * ND * 2;
    unsigned short* Wvb = (unsigned short*)ws;  ws += (size_t)ND * ND * 2;
    unsigned short* Wob = (unsigned short*)ws;  ws += (size_t)ND * ND * 2;
    float*          Qf  = (float*)ws;           ws += (size_t)NT * ND * 4;
    float*          Kf  = (float*)ws;           ws += (size_t)NT * ND * 4;
    unsigned short* Qbf = (unsigned short*)ws;  ws += (size_t)NT * ND * 2;
    unsigned short* Kbf = (unsigned short*)ws;  ws += (size_t)NT * ND * 2;
    unsigned short* Vt  = (unsigned short*)ws;  ws += (size_t)NT * ND * 2;
    unsigned short* Ctx = (unsigned short*)ws;  ws += (size_t)NT * ND * 2;

    // 1) fp32 -> bf16
    k_f32_to_bf16<<<2048, 256, 0, stream>>>(hidden, Xbf, NT * ND);
    k_f32_to_bf16<<<1024, 256, 0, stream>>>(wq, Wqb, ND * ND);
    k_f32_to_bf16<<<1024, 256, 0, stream>>>(wk, Wkb, ND * ND);
    k_f32_to_bf16<<<1024, 256, 0, stream>>>(wv, Wvb, ND * ND);
    k_f32_to_bf16<<<1024, 256, 0, stream>>>(wo, Wob, ND * ND);

    // 2) Q/K/V projections (WMMA, 32x64/wave).  V: f32 + bf16 transposed.
    k_gemm_bf16<<<1024, 128, 0, stream>>>(Xbf, Wqb, Qf, nullptr, 0);
    k_gemm_bf16<<<1024, 128, 0, stream>>>(Xbf, Wkb, Kf, nullptr, 0);
    k_gemm_bf16<<<1024, 128, 0, stream>>>(Xbf, Wvb, vout, Vt, 1);

    // 3) RoPE (also produces f32 K output)
    k_rope<<<16384, 256, 0, stream>>>(Qf, Kf, cosT, sinT, Qbf, kout, Kbf);

    // 4) causal flash attention (WMMA + TDM/async staging)
    k_flash_attn<<<1024, 128, 0, stream>>>(Qbf, Kbf, Vt, Ctx);

    // 5) output projection (WMMA), f32 row-major into d_out
    k_gemm_bf16<<<1024, 128, 0, stream>>>(Ctx, Wob, out, nullptr, 2);
}
